// DecomposeNet_36747740185146
// MI455X (gfx1250) — compile-verified, matrix operand order checked by reading, not run on hardware
//
#include <hip/hip_runtime.h>

typedef __attribute__((ext_vector_type(16))) _Float16 v16h;
typedef __attribute__((ext_vector_type(8)))  float    v8f;

#define EPSV 1e-5f

// ---------------------------------------------------------------------------
// Haar bands: cur (N,C,2h,2w) -> bands (N,4C,h,w), channel = c*4 + band
// ---------------------------------------------------------------------------
__global__ void haar_kernel(const float* __restrict__ in, float* __restrict__ out,
                            int N, int C, int h, int w) {
  int i = blockIdx.x * blockDim.x + threadIdx.x;
  int total = N * C * 4 * h * w;
  if (i >= total) return;
  int x = i % w;
  int y = (i / w) % h;
  int ch = (i / (w * h)) % (4 * C);
  int n = i / (w * h * 4 * C);
  int c = ch >> 2, band = ch & 3;
  const float* p = in + (size_t)(n * C + c) * (2 * h) * (2 * w);
  int W2 = 2 * w;
  float a  = p[(2 * y) * W2 + 2 * x];
  float b  = p[(2 * y) * W2 + 2 * x + 1];
  float cc = p[(2 * y + 1) * W2 + 2 * x];
  float d  = p[(2 * y + 1) * W2 + 2 * x + 1];
  float v;
  if      (band == 0) v = (a + b + cc + d) * 0.5f;
  else if (band == 1) v = (a + b - cc - d) * 0.5f;
  else if (band == 2) v = (a - b + cc - d) * 0.5f;
  else                v = (a - b - cc + d) * 0.5f;
  out[i] = v;
}

// ---------------------------------------------------------------------------
// Bilinear resize (half-pixel centers, edge clamp) into a channel slice.
// ---------------------------------------------------------------------------
__global__ void resize_bilinear_kernel(const float* __restrict__ in, float* __restrict__ out,
                                       int N, int C, int ih, int iw, int oh, int ow,
                                       int outCtot, int co_off) {
  int i = blockIdx.x * blockDim.x + threadIdx.x;
  int total = N * C * oh * ow;
  if (i >= total) return;
  int x = i % ow;
  int y = (i / ow) % oh;
  int c = (i / (ow * oh)) % C;
  int n = i / (ow * oh * C);
  float sy = (y + 0.5f) * (float)ih / (float)oh - 0.5f;
  float sx = (x + 0.5f) * (float)iw / (float)ow - 0.5f;
  if (sy < 0.f) sy = 0.f; if (sy > (float)(ih - 1)) sy = (float)(ih - 1);
  if (sx < 0.f) sx = 0.f; if (sx > (float)(iw - 1)) sx = (float)(iw - 1);
  int y0 = (int)sy, x0 = (int)sx;
  float fy = sy - y0, fx = sx - x0;
  int y1 = y0 + 1 < ih ? y0 + 1 : ih - 1;
  int x1 = x0 + 1 < iw ? x0 + 1 : iw - 1;
  const float* p = in + (size_t)(n * C + c) * ih * iw;
  float v00 = p[y0 * iw + x0], v01 = p[y0 * iw + x1];
  float v10 = p[y1 * iw + x0], v11 = p[y1 * iw + x1];
  float v = v00 * (1 - fy) * (1 - fx) + v01 * (1 - fy) * fx
          + v10 * fy * (1 - fx)       + v11 * fy * fx;
  out[((size_t)(n * outCtot + co_off + c) * oh + y) * ow + x] = v;
}

// ---------------------------------------------------------------------------
// 2x2 average pool
// ---------------------------------------------------------------------------
__global__ void avgpool2_kernel(const float* __restrict__ in, float* __restrict__ out,
                                int N, int C, int h, int w) {
  int i = blockIdx.x * blockDim.x + threadIdx.x;
  int total = N * C * h * w;
  if (i >= total) return;
  int x = i % w;
  int y = (i / w) % h;
  int nc = i / (w * h);
  const float* p = in + (size_t)nc * (2 * h) * (2 * w);
  int W2 = 2 * w;
  out[i] = 0.25f * (p[(2 * y) * W2 + 2 * x] + p[(2 * y) * W2 + 2 * x + 1] +
                    p[(2 * y + 1) * W2 + 2 * x] + p[(2 * y + 1) * W2 + 2 * x + 1]);
}

// ---------------------------------------------------------------------------
// Pack OIHW fp32 weights -> f16 [CoutPad][Kpad], zero-padded
// ---------------------------------------------------------------------------
__global__ void pack_w_kernel(const float* __restrict__ w, _Float16* __restrict__ wp,
                              int Cout, int K, int Kpad, int CoutPad) {
  int i = blockIdx.x * blockDim.x + threadIdx.x;
  int total = CoutPad * Kpad;
  if (i >= total) return;
  int co = i / Kpad, k = i % Kpad;
  wp[i] = (co < Cout && k < K) ? (_Float16)w[co * K + k] : (_Float16)0.f;
}

// ---------------------------------------------------------------------------
// Implicit-GEMM conv with WMMA f32_16x16x32_f16.
// One wave per block; wave computes 32 pixels (two 16-pixel M tiles sharing
// the B fragments) x NB*16 output channels. Grid = (W/32, H, N): no integer
// division in the kernel. A-patch (3x34 halo x Cin, or 32 x Cin) staged in
// LDS as f16; B = prepacked f16 weights [nb*16+m][Kpad].
// ---------------------------------------------------------------------------
template <int IS3X3, int NB>
__global__ void conv_wmma_kernel(const float* __restrict__ in,
                                 const _Float16* __restrict__ wp,
                                 const float* __restrict__ bias,
                                 float* __restrict__ out,
                                 int N, int Cin, int H, int W, int K, int Kpad,
                                 int Cout, int CoutTot, int co_off) {
  __shared__ _Float16 lds[IS3X3 ? 64 * 102 : 64 * 32];
  int x0 = blockIdx.x << 5;   // 32 output pixels along the row
  int y  = blockIdx.y;
  int n  = blockIdx.z;
  int lane = threadIdx.x;

  // Stage the f16 input patch in LDS.
  int tot = IS3X3 ? Cin * 102 : Cin * 32;
  for (int e = lane; e < tot; e += 32) {
    float v;
    if (IS3X3) {
      int ci = e / 102;
      int rm = e - ci * 102;
      int r  = rm / 34;
      int cc = rm - r * 34;
      int gy = y - 1 + r;
      int gx = x0 - 1 + cc;
      v = (gy >= 0 && gy < H && gx >= 0 && gx < W)
            ? in[((size_t)(n * Cin + ci) * H + gy) * W + gx] : 0.f;
    } else {
      int ci = e >> 5;
      int m2 = e & 31;
      v = in[((size_t)(n * Cin + ci) * H + y) * W + x0 + m2];
    }
    lds[e] = (_Float16)v;
  }
  __syncthreads();

  int hi = lane >> 4;     // 0: lanes 0-15, 1: lanes 16-31
  int m  = lane & 15;     // pixel-in-segment for A, out-channel-in-block for B

  v8f acc[2][NB] = {};
  int ksteps = Kpad >> 5;
  for (int kb = 0; kb < ksteps; ++kb) {
    int kbase = (kb << 5) + hi * 8;

    // B fragments (one per out-channel block), reused by both M segments.
    v16h b[NB];
#pragma unroll
    for (int nb = 0; nb < NB; ++nb) {
      const _Float16* wr = wp + (size_t)(nb * 16 + m) * Kpad + kbase;
#pragma unroll
      for (int i2 = 0; i2 < 16; ++i2) b[nb][i2] = wr[i2 < 8 ? i2 : i2 + 8];
    }

#pragma unroll
    for (int s = 0; s < 2; ++s) {
      // A fragment: 16-bit A 16x32 layout
      // (lane<16: K base+0..7,16..23; lane>=16: base+8..15,24..31)
      v16h a;
#pragma unroll
      for (int i2 = 0; i2 < 16; ++i2) {
        int k = kbase + (i2 < 8 ? i2 : i2 + 8);
        _Float16 v = (_Float16)0.f;
        if (k < K) {
          if (IS3X3) {
            int ci = k / 9;
            int rm = k - ci * 9;
            v = lds[ci * 102 + (rm / 3) * 34 + (s * 16 + m) + (rm % 3)];
          } else {
            v = lds[(k << 5) + s * 16 + m];
          }
        }
        a[i2] = v;
      }
#pragma unroll
      for (int nb = 0; nb < NB; ++nb) {
        acc[s][nb] = __builtin_amdgcn_wmma_f32_16x16x32_f16(
            false, a, false, b[nb], (short)0, acc[s][nb], false, false);
      }
    }
  }

  // D layout: lane -> N = m; VGPR r -> pixel M = r + hi*8 (within segment s)
#pragma unroll
  for (int s = 0; s < 2; ++s) {
#pragma unroll
    for (int nb = 0; nb < NB; ++nb) {
      int co = nb * 16 + m;
      if (co < Cout) {
        float bz = bias[co];
        float* op = out + ((size_t)(n * CoutTot + co_off + co) * H + y) * W
                        + x0 + s * 16 + hi * 8;
#pragma unroll
        for (int r = 0; r < 8; ++r) op[r] = acc[s][nb][r] + bz;
      }
    }
  }
}

// ---------------------------------------------------------------------------
// Instance norm + PReLU in place over a channel slice: per (n,c) over HW.
// ---------------------------------------------------------------------------
__global__ void inorm_prelu_kernel(float* __restrict__ p, int Ctot, int co_off,
                                   int C, int HW, const float* __restrict__ alpha) {
  int c = blockIdx.x % C;
  int n = blockIdx.x / C;
  float* base = p + (size_t)(n * Ctot + co_off + c) * HW;
  __shared__ float s1[256], s2[256];
  float sum = 0.f, sq = 0.f;
  for (int i = threadIdx.x; i < HW; i += 256) {
    float v = base[i];
    sum += v; sq += v * v;
  }
  s1[threadIdx.x] = sum; s2[threadIdx.x] = sq;
  __syncthreads();
  for (int s = 128; s > 0; s >>= 1) {
    if (threadIdx.x < s) { s1[threadIdx.x] += s1[threadIdx.x + s]; s2[threadIdx.x] += s2[threadIdx.x + s]; }
    __syncthreads();
  }
  float mean = s1[0] / (float)HW;
  float var  = s2[0] / (float)HW - mean * mean;
  float rs = rsqrtf(var + EPSV);
  float al = alpha[0];
  for (int i = threadIdx.x; i < HW; i += 256) {
    float v = (base[i] - mean) * rs;
    base[i] = v > 0.f ? v : al * v;
  }
}

// ---------------------------------------------------------------------------
// Batch-norm stats per channel over (N, HW) — nested loops, no div in the
// hot path.
// ---------------------------------------------------------------------------
__global__ void bn_stats_kernel(const float* __restrict__ p, float* __restrict__ mean,
                                float* __restrict__ var, int N, int C, int HW) {
  int c = blockIdx.x;
  __shared__ float s1[256], s2[256];
  float sum = 0.f, sq = 0.f;
  for (int n = 0; n < N; ++n) {
    const float* base = p + ((size_t)n * C + c) * HW;
    for (int i = threadIdx.x; i < HW; i += 256) {
      float v = base[i];
      sum += v; sq += v * v;
    }
  }
  s1[threadIdx.x] = sum; s2[threadIdx.x] = sq;
  __syncthreads();
  for (int s = 128; s > 0; s >>= 1) {
    if (threadIdx.x < s) { s1[threadIdx.x] += s1[threadIdx.x + s]; s2[threadIdx.x] += s2[threadIdx.x + s]; }
    __syncthreads();
  }
  if (threadIdx.x == 0) {
    float inv = 1.f / ((float)N * (float)HW);
    float mval = s1[0] * inv;
    mean[c] = mval;
    var[c]  = s2[0] * inv - mval * mval;
  }
}

// hwShift: log2(HW) — all spatial sizes here are powers of two.
__global__ void bn_prelu_kernel(float* __restrict__ p, const float* __restrict__ mean,
                                const float* __restrict__ var, const float* __restrict__ g,
                                const float* __restrict__ b, const float* __restrict__ alpha,
                                int C, int hwShift, int total) {
  int i = blockIdx.x * blockDim.x + threadIdx.x;
  if (i >= total) return;
  int c = (i >> hwShift) % C;
  float v = (p[i] - mean[c]) * rsqrtf(var[c] + EPSV) * g[c] + b[c];
  p[i] = v > 0.f ? v : alpha[0] * v;
}

// ---------------------------------------------------------------------------
// Channel attention pieces
// ---------------------------------------------------------------------------
__global__ void pool_avgmax_kernel(const float* __restrict__ p, float* __restrict__ avg,
                                   float* __restrict__ mx, int C, int HW) {
  int c = blockIdx.x % C;
  int n = blockIdx.x / C;
  const float* base = p + (size_t)(n * C + c) * HW;
  __shared__ float s1[256], s2[256];
  float sum = 0.f, mval = -3.4e38f;
  for (int i = threadIdx.x; i < HW; i += 256) {
    float v = base[i];
    sum += v; mval = fmaxf(mval, v);
  }
  s1[threadIdx.x] = sum; s2[threadIdx.x] = mval;
  __syncthreads();
  for (int s = 128; s > 0; s >>= 1) {
    if (threadIdx.x < s) {
      s1[threadIdx.x] += s1[threadIdx.x + s];
      s2[threadIdx.x] = fmaxf(s2[threadIdx.x], s2[threadIdx.x + s]);
    }
    __syncthreads();
  }
  if (threadIdx.x == 0) {
    avg[n * C + c] = s1[0] / (float)HW;
    mx[n * C + c]  = s2[0];
  }
}

__global__ void attn_fc_kernel(const float* __restrict__ avg, const float* __restrict__ mx,
                               const float* __restrict__ w1, const float* __restrict__ w2,
                               float* __restrict__ scale) {
  int n = blockIdx.x;
  int t = threadIdx.x; // blockDim = 64
  __shared__ float ha[16], hm[16];
  if (t < 16) {
    float s = 0.f;
    for (int c = 0; c < 64; ++c) s += w1[t * 64 + c] * avg[n * 64 + c];
    ha[t] = fmaxf(s, 0.f);
  } else if (t < 32) {
    int h = t - 16;
    float s = 0.f;
    for (int c = 0; c < 64; ++c) s += w1[h * 64 + c] * mx[n * 64 + c];
    hm[h] = fmaxf(s, 0.f);
  }
  __syncthreads();
  float s1 = 0.f, s2 = 0.f;
  for (int h = 0; h < 16; ++h) {
    s1 += w2[t * 16 + h] * ha[h];
    s2 += w2[t * 16 + h] * hm[h];
  }
  scale[n * 64 + t] = 1.f / (1.f + expf(-s1)) + 1.f / (1.f + expf(-s2));
}

__global__ void scale_mul_kernel(const float* __restrict__ in, const float* __restrict__ sc,
                                 float* __restrict__ out, int C, int hwShift, int total) {
  int i = blockIdx.x * blockDim.x + threadIdx.x;
  if (i >= total) return;
  int nc = i >> hwShift;
  int c = nc % C;
  int n = nc / C;
  out[i] = in[i] * sc[n * C + c];
}

// ---------------------------------------------------------------------------
// Final BN + sigmoid, splitting (N,6,HW) into out0=(N,3,HW) then out1=(N,3,HW)
// ---------------------------------------------------------------------------
__global__ void bn_sigmoid_out_kernel(const float* __restrict__ p, const float* __restrict__ mean,
                                      const float* __restrict__ var, const float* __restrict__ g,
                                      const float* __restrict__ b, float* __restrict__ out,
                                      int N, int C, int hwShift, int HW) {
  int i = blockIdx.x * blockDim.x + threadIdx.x;
  int total = N * C * HW;
  if (i >= total) return;
  int off = i & (HW - 1);
  int nc = i >> hwShift;
  int c = nc % C;
  int n = nc / C;
  float v = (p[i] - mean[c]) * rsqrtf(var[c] + EPSV) * g[c] + b[c];
  v = 1.f / (1.f + expf(-v));
  size_t HALF = (size_t)N * 3 * HW;
  size_t dst = (c < 3) ? ((size_t)(n * 3 + c) * HW + off)
                       : (HALF + (size_t)(n * 3 + (c - 3)) * HW + off);
  out[dst] = v;
}

// ---------------------------------------------------------------------------
// Orchestration
// ---------------------------------------------------------------------------
static inline int cdiv(int a, int b) { return (a + b - 1) / b; }

extern "C" void kernel_launch(void* const* d_in, const int* in_sizes, int n_in,
                              void* d_out, int out_size, void* d_ws, size_t ws_size,
                              hipStream_t stream) {
  const float* x      = (const float*)d_in[0];
  const float* w_wav0 = (const float*)d_in[1];
  const float* b_wav0 = (const float*)d_in[2];
  const float* a_wav0 = (const float*)d_in[3];
  const float* w_wav1 = (const float*)d_in[4];
  const float* b_wav1 = (const float*)d_in[5];
  const float* a_wav1 = (const float*)d_in[6];
  const float* w_wav2 = (const float*)d_in[7];
  const float* b_wav2 = (const float*)d_in[8];
  const float* a_wav2 = (const float*)d_in[9];
  const float* w_fus  = (const float*)d_in[10];
  const float* b_fus  = (const float*)d_in[11];
  const float* a_fus  = (const float*)d_in[12];
  const float* w_fa1  = (const float*)d_in[13];
  const float* w_fa2  = (const float*)d_in[14];
  const float* w_fe1  = (const float*)d_in[15];
  const float* b_fe1  = (const float*)d_in[16];
  const float* g_bn1  = (const float*)d_in[17];
  const float* be_bn1 = (const float*)d_in[18];
  const float* a_fe1  = (const float*)d_in[19];
  const float* w_fe2  = (const float*)d_in[20];
  const float* b_fe2  = (const float*)d_in[21];
  const float* g_bn2  = (const float*)d_in[22];
  const float* be_bn2 = (const float*)d_in[23];
  const float* a_fe2  = (const float*)d_in[24];
  const float* w_sep  = (const float*)d_in[25];
  const float* b_sep  = (const float*)d_in[26];
  const float* g_bn3  = (const float*)d_in[27];
  const float* be_bn3 = (const float*)d_in[28];
  float* outp = (float*)d_out;

  const int S = 256 * 256;
  float* ws = (float*)d_ws;
  float* fA        = ws;                                    // (8,64,256,256)
  float* fB        = fA + (size_t)8 * 64 * S;               // (8,64,256,256)
  float* feats     = fB + (size_t)8 * 64 * S;               // (8,63,256,256)
  float* bands     = feats + (size_t)8 * 63 * S;            // up to (8,12,256,256); also sep out
  float* bandsHalf = bands + (size_t)8 * 12 * S;            // up to (8,12,128,128)
  float* cur1      = bandsHalf + (size_t)8 * 12 * 16384;    // (8,3,128,128)
  float* cur2      = cur1 + (size_t)8 * 3 * 16384;          // (8,3,64,64)
  float* featLow   = cur2 + (size_t)8 * 3 * 4096;           // up to (8,21,128,128)
  _Float16* wpack  = (_Float16*)(featLow + (size_t)8 * 21 * 16384); // 64x640 f16
  float* stats = (float*)(wpack + 64 * 640);
  float* meanb  = stats;         // 64
  float* varb   = stats + 64;    // 64
  float* avgb   = stats + 128;   // 512
  float* mxb    = avgb + 512;    // 512
  float* scaleb = mxb + 512;     // 512

  // ---------------- level 0 (256x256) ----------------
  haar_kernel<<<cdiv(8 * 12 * 128 * 128, 256), 256, 0, stream>>>(x, bandsHalf, 8, 3, 128, 128);
  resize_bilinear_kernel<<<cdiv(8 * 12 * S, 256), 256, 0, stream>>>(bandsHalf, bands, 8, 12, 128, 128, 256, 256, 12, 0);
  pack_w_kernel<<<cdiv(32 * 128, 256), 256, 0, stream>>>(w_wav0, wpack, 21, 108, 128, 32);
  conv_wmma_kernel<1, 2><<<dim3(8, 256, 8), 32, 0, stream>>>(bands, wpack, b_wav0, feats,
                                                             8, 12, 256, 256, 108, 128, 21, 63, 0);
  inorm_prelu_kernel<<<8 * 21, 256, 0, stream>>>(feats, 63, 0, 21, S, a_wav0);

  // ---------------- level 1 (128x128) ----------------
  avgpool2_kernel<<<cdiv(8 * 3 * 128 * 128, 256), 256, 0, stream>>>(x, cur1, 8, 3, 128, 128);
  haar_kernel<<<cdiv(8 * 12 * 64 * 64, 256), 256, 0, stream>>>(cur1, bandsHalf, 8, 3, 64, 64);
  resize_bilinear_kernel<<<cdiv(8 * 12 * 128 * 128, 256), 256, 0, stream>>>(bandsHalf, bands, 8, 12, 64, 64, 128, 128, 12, 0);
  pack_w_kernel<<<cdiv(32 * 128, 256), 256, 0, stream>>>(w_wav1, wpack, 21, 108, 128, 32);
  conv_wmma_kernel<1, 2><<<dim3(4, 128, 8), 32, 0, stream>>>(bands, wpack, b_wav1, featLow,
                                                             8, 12, 128, 128, 108, 128, 21, 21, 0);
  inorm_prelu_kernel<<<8 * 21, 256, 0, stream>>>(featLow, 21, 0, 21, 16384, a_wav1);
  resize_bilinear_kernel<<<cdiv(8 * 21 * S, 256), 256, 0, stream>>>(featLow, feats, 8, 21, 128, 128, 256, 256, 63, 21);

  // ---------------- level 2 (64x64) ----------------
  avgpool2_kernel<<<cdiv(8 * 3 * 64 * 64, 256), 256, 0, stream>>>(cur1, cur2, 8, 3, 64, 64);
  haar_kernel<<<cdiv(8 * 12 * 32 * 32, 256), 256, 0, stream>>>(cur2, bandsHalf, 8, 3, 32, 32);
  resize_bilinear_kernel<<<cdiv(8 * 12 * 64 * 64, 256), 256, 0, stream>>>(bandsHalf, bands, 8, 12, 32, 32, 64, 64, 12, 0);
  pack_w_kernel<<<cdiv(32 * 128, 256), 256, 0, stream>>>(w_wav2, wpack, 21, 108, 128, 32);
  conv_wmma_kernel<1, 2><<<dim3(2, 64, 8), 32, 0, stream>>>(bands, wpack, b_wav2, featLow,
                                                            8, 12, 64, 64, 108, 128, 21, 21, 0);
  inorm_prelu_kernel<<<8 * 21, 256, 0, stream>>>(featLow, 21, 0, 21, 4096, a_wav2);
  resize_bilinear_kernel<<<cdiv(8 * 21 * S, 256), 256, 0, stream>>>(featLow, feats, 8, 21, 64, 64, 256, 256, 63, 42);

  // ---------------- fuse: 1x1 conv 63 -> 64, inorm + prelu ----------------
  pack_w_kernel<<<cdiv(64 * 64, 256), 256, 0, stream>>>(w_fus, wpack, 64, 63, 64, 64);
  conv_wmma_kernel<0, 4><<<dim3(8, 256, 8), 32, 0, stream>>>(feats, wpack, b_fus, fA,
                                                             8, 63, 256, 256, 63, 64, 64, 64, 0);
  inorm_prelu_kernel<<<8 * 64, 256, 0, stream>>>(fA, 64, 0, 64, S, a_fus);

  // ---------------- channel attention ----------------
  pool_avgmax_kernel<<<8 * 64, 256, 0, stream>>>(fA, avgb, mxb, 64, S);
  attn_fc_kernel<<<8, 64, 0, stream>>>(avgb, mxb, w_fa1, w_fa2, scaleb);
  scale_mul_kernel<<<cdiv(8 * 64 * S, 256), 256, 0, stream>>>(fA, scaleb, fB, 64, 16, 8 * 64 * S);

  // ---------------- fe1: 3x3 conv 64->64, BN + prelu ----------------
  pack_w_kernel<<<cdiv(64 * 576, 256), 256, 0, stream>>>(w_fe1, wpack, 64, 576, 576, 64);
  conv_wmma_kernel<1, 4><<<dim3(8, 256, 8), 32, 0, stream>>>(fB, wpack, b_fe1, fA,
                                                             8, 64, 256, 256, 576, 576, 64, 64, 0);
  bn_stats_kernel<<<64, 256, 0, stream>>>(fA, meanb, varb, 8, 64, S);
  bn_prelu_kernel<<<cdiv(8 * 64 * S, 256), 256, 0, stream>>>(fA, meanb, varb, g_bn1, be_bn1, a_fe1, 64, 16, 8 * 64 * S);

  // ---------------- fe2: 3x3 conv 64->64, BN + prelu ----------------
  pack_w_kernel<<<cdiv(64 * 576, 256), 256, 0, stream>>>(w_fe2, wpack, 64, 576, 576, 64);
  conv_wmma_kernel<1, 4><<<dim3(8, 256, 8), 32, 0, stream>>>(fA, wpack, b_fe2, fB,
                                                             8, 64, 256, 256, 576, 576, 64, 64, 0);
  bn_stats_kernel<<<64, 256, 0, stream>>>(fB, meanb, varb, 8, 64, S);
  bn_prelu_kernel<<<cdiv(8 * 64 * S, 256), 256, 0, stream>>>(fB, meanb, varb, g_bn2, be_bn2, a_fe2, 64, 16, 8 * 64 * S);

  // ---------------- sep: 3x3 conv 64->6, BN + sigmoid -> d_out ----------------
  pack_w_kernel<<<cdiv(16 * 576, 256), 256, 0, stream>>>(w_sep, wpack, 6, 576, 576, 16);
  conv_wmma_kernel<1, 1><<<dim3(8, 256, 8), 32, 0, stream>>>(fB, wpack, b_sep, bands,
                                                             8, 64, 256, 256, 576, 576, 6, 6, 0);
  bn_stats_kernel<<<6, 256, 0, stream>>>(bands, meanb, varb, 8, 6, S);
  bn_sigmoid_out_kernel<<<cdiv(8 * 6 * S, 256), 256, 0, stream>>>(bands, meanb, varb, g_bn3, be_bn3, outp, 8, 6, 16, S);
}
